// HungarianMatcher_25950192403140
// MI455X (gfx1250) — compile-verified, matrix operand order checked by reading, not run on hardware
//
#include <hip/hip_runtime.h>
#include <hip/hip_bf16.h>
#include <math.h>

typedef __attribute__((ext_vector_type(16))) _Float16 v16h;
typedef __attribute__((ext_vector_type(8)))  float    v8f;

// Exact pointee type of the async-LDS builtins: GCC-style vector of 4 ints.
typedef int async_v4i __attribute__((vector_size(4 * sizeof(int))));
typedef __attribute__((address_space(1))) async_v4i* async_gptr;
typedef __attribute__((address_space(3))) async_v4i* async_lptr;

#define B_   64
#define N_   128
#define C_   128
#define S_   1024
#define G_   24
#define NROWS (B_ * N_)            // 8192 rows per tensor
#define COST_ELEMS (B_ * N_ * G_)  // 196608

// ---- CDNA5 async global->LDS support (guarded; falls back to manual copy) --
#if defined(__has_builtin)
#if __has_builtin(__builtin_amdgcn_global_load_async_to_lds_b128)
#define HAVE_ASYNC_LDS 1
#endif
#if __has_builtin(__builtin_amdgcn_s_wait_asynccnt)
#define HAVE_WAIT_ASYNC_BUILTIN 1
#endif
#endif

__device__ __forceinline__ void wait_async_zero() {
#if defined(HAVE_WAIT_ASYNC_BUILTIN)
    __builtin_amdgcn_s_wait_asynccnt(0);
#elif defined(HAVE_ASYNC_LDS)
    asm volatile("s_wait_asynccnt 0x0" ::: "memory");
#endif
}

// ---------------------------------------------------------------------------
// Kernel 1: per-row (max, sum(exp(x - max))) for 5 tensors.
//   blockIdx.y: 0=ent_head_0, 1=ent_end_0, 2=ent_head_1, 3=ent_end_1 (S=1024),
//               4=pred_rel (S=128).
//   One wave (32 threads) handles 16 consecutive rows staged in LDS via
//   GLOBAL_LOAD_ASYNC_TO_LDS_B128 (ASYNCcnt path), then sums exp() rows with
//   v_wmma_f32_16x16x32_f16 against an all-ones B (two-accumulator unroll).
// ---------------------------------------------------------------------------
__global__ __launch_bounds__(32) void row_stats_kernel(
    const float* __restrict__ rel,
    const float* __restrict__ h0, const float* __restrict__ e0,
    const float* __restrict__ h1, const float* __restrict__ e1,
    float* __restrict__ smax, float* __restrict__ ssum)
{
    __shared__ __align__(16) float lds[16 * S_];   // 64 KB

    const int t = blockIdx.y;
    const int S = (t == 4) ? C_ : S_;
    const float* src = (t == 0) ? h0 : (t == 1) ? e0 : (t == 2) ? h1
                     : (t == 3) ? e1 : rel;
    const int rowBase = blockIdx.x * 16;
    const float* base = src + (size_t)rowBase * S;   // 16 rows are contiguous
    const int lane = threadIdx.x;

    // Stage 16*S floats into LDS (uniform trip count => EXEC stays all-ones).
    const int total4 = 16 * S / 4;
    const float4* g4 = (const float4*)base;
    float4* l4 = (float4*)lds;
#if defined(HAVE_ASYNC_LDS)
    for (int i = lane; i < total4; i += 32) {
        __builtin_amdgcn_global_load_async_to_lds_b128(
            (async_gptr)(g4 + i), (async_lptr)(l4 + i),
            /*offset=*/0, /*cpol=*/0);
    }
    wait_async_zero();
#else
    for (int i = lane; i < total4; i += 32) l4[i] = g4[i];
#endif
    __syncthreads();

    const int row  = lane & 15;    // logical row this lane serves
    const int half = lane >> 4;    // which half of the row / chunk
    const int hlen = S >> 1;

    // Row max: lane pair (l, l+16) each scans half the row, combine via shuffle.
    float m = -INFINITY;
    const float* rp = lds + row * S + half * hlen;
    for (int k = 0; k < hlen; k += 4) {
        float4 v = *(const float4*)(rp + k);
        m = fmaxf(m, fmaxf(fmaxf(v.x, v.y), fmaxf(v.z, v.w)));
    }
    m = fmaxf(m, __shfl_xor(m, 16, 32));

    // Sum of exp via WMMA: A[16x32] = exp chunk (f16), B = ones.
    // Lane l holds 16 elements of row (l&15); union with lane l+16 covers the
    // 32-wide K chunk, so D[r][*] = row-r chunk sum regardless of K placement.
    v16h ones;
    #pragma unroll
    for (int i = 0; i < 16; ++i) ones[i] = (_Float16)1.0f;

    v8f acc0 = {};
    v8f acc1 = {};
    const float* cp = lds + row * S + half * 16;
    const int nchunks = S / 32;          // 32 (ent) or 4 (rel), always even
    for (int c = 0; c < nchunks; c += 2) {
        const float* p0 = cp + (c + 0) * 32;
        const float* p1 = cp + (c + 1) * 32;
        v16h a0, a1;
        #pragma unroll
        for (int i = 0; i < 16; ++i) a0[i] = (_Float16)__expf(p0[i] - m);
        #pragma unroll
        for (int i = 0; i < 16; ++i) a1[i] = (_Float16)__expf(p1[i] - m);
        acc0 = __builtin_amdgcn_wmma_f32_16x16x32_f16(
            false, a0, false, ones, (short)0, acc0, false, false);
        acc1 = __builtin_amdgcn_wmma_f32_16x16x32_f16(
            false, a1, false, ones, (short)0, acc1, false, false);
    }

    // C/D layout: lanes 0-15 VGPR r = D[r][lane]; lanes 16-31 VGPR r = D[r+8][lane-16].
    const size_t srow = (size_t)t * NROWS + rowBase;
    if (lane < 16) smax[srow + lane] = m;
    if (lane == 0) {
        #pragma unroll
        for (int r = 0; r < 8; ++r) ssum[srow + r] = acc0[r] + acc1[r];
    }
    if (lane == 16) {
        #pragma unroll
        for (int r = 0; r < 8; ++r) ssum[srow + 8 + r] = acc0[r] + acc1[r];
    }
}

// ---------------------------------------------------------------------------
// Kernel 2: cost[b][n][g] from gathered logits + row stats.
// ---------------------------------------------------------------------------
__global__ __launch_bounds__(256) void cost_kernel(
    const float* __restrict__ rel,
    const float* __restrict__ h0, const float* __restrict__ e0,
    const float* __restrict__ h1, const float* __restrict__ e1,
    const int* __restrict__ gold_rel, const int* __restrict__ gold_ent,
    const float* __restrict__ smax, const float* __restrict__ ssum,
    float* __restrict__ out)
{
    int idx = blockIdx.x * blockDim.x + threadIdx.x;
    if (idx >= COST_ELEMS) return;
    const int g = idx % G_;
    const int n = (idx / G_) % N_;
    const int b = idx / (G_ * N_);
    const int row = b * N_ + n;

    // rel (stats tensor 4)
    const int gr = gold_rel[b * G_ + g];
    float c = -__expf(rel[(size_t)row * C_ + gr] - smax[4 * NROWS + row])
              / ssum[4 * NROWS + row];

    const float* H[2] = { h0, h1 };
    const float* E[2] = { e0, e1 };
    #pragma unroll
    for (int i = 0; i < 2; ++i) {
        const int gh = gold_ent[((b * G_ + g) * 2 + i) * 2 + 0];
        const int ge = gold_ent[((b * G_ + g) * 2 + i) * 2 + 1];
        const int th = 2 * i, te = 2 * i + 1;
        float ph = __expf(H[i][(size_t)row * S_ + gh] - smax[th * NROWS + row])
                   / ssum[th * NROWS + row];
        float pe = __expf(E[i][(size_t)row * S_ + ge] - smax[te * NROWS + row])
                   / ssum[te * NROWS + row];
        c -= 0.5f * (ph + pe);
    }
    out[idx] = c;   // idx == b*N*G + n*G + g (row-major B,N,G)
}

// ---------------------------------------------------------------------------
// Kernel 3: Jonker-Volgenant LSA on cost[b].T (24 rows x 128 cols) per batch.
// One block per batch, thread t owns column j = t+1. f64 to match reference.
// ---------------------------------------------------------------------------
__global__ __launch_bounds__(128) void lsa_kernel(
    const float* __restrict__ cost, float* __restrict__ row_out,
    float* __restrict__ col_out)
{
    const int n = G_;      // 24 rows (gold)
    const int b = blockIdx.x;
    const int t = threadIdx.x;
    const int j = t + 1;

    __shared__ double u[G_ + 1];
    __shared__ double v[N_ + 1];
    __shared__ double minv[N_ + 1];
    __shared__ int way[N_ + 1];
    __shared__ int p[N_ + 1];
    __shared__ int usedArr[N_ + 1];
    __shared__ double redv[N_];
    __shared__ int redi[N_];
    __shared__ int sh_j0;
    __shared__ int colforrow[G_];

    // Cache this thread's cost column: cT[i][j-1] = cost[b][t][i]
    float cc[G_];
    #pragma unroll
    for (int i = 0; i < G_; ++i)
        cc[i] = cost[((size_t)b * N_ + t) * G_ + i];

    if (t <= n) u[t] = 0.0;
    v[j] = 0.0;  p[j] = 0;  way[j] = 0;
    if (t == 0) { v[0] = 0.0; p[0] = 0; way[0] = 0; }
    __syncthreads();

    for (int i = 1; i <= n; ++i) {
        if (t == 0) { p[0] = i; sh_j0 = 0; }
        minv[j] = INFINITY;
        usedArr[j] = 0;
        if (t == 0) usedArr[0] = 0;
        __syncthreads();

        while (true) {
            const int j0 = sh_j0;
            if (t == 0) usedArr[j0] = 1;
            __syncthreads();

            const int i0 = p[j0];
            if (!usedArr[j]) {
                double cur = (double)cc[i0 - 1] - u[i0] - v[j];
                if (cur < minv[j]) { minv[j] = cur; way[j] = j0; }
            }
            redv[t] = usedArr[j] ? INFINITY : minv[j];
            redi[t] = j;
            __syncthreads();

            // argmin with first-index tie-break (matches numpy argmin)
            for (int s = 64; s > 0; s >>= 1) {
                if (t < s) {
                    double ov = redv[t + s]; int oi = redi[t + s];
                    if (ov < redv[t] || (ov == redv[t] && oi < redi[t])) {
                        redv[t] = ov; redi[t] = oi;
                    }
                }
                __syncthreads();
            }
            const double delta = redv[0];
            const int j1 = redi[0];

            // dual updates (distinct p[j] rows for used cols => no conflicts)
            if (usedArr[j]) { u[p[j]] += delta; v[j] -= delta; }
            else            { minv[j] -= delta; }
            if (t == 0) { u[p[0]] += delta; sh_j0 = j1; }
            __syncthreads();

            if (p[j1] == 0) break;
        }

        if (t == 0) {  // augment along alternating path
            int j0 = sh_j0;
            while (j0) { int jn = way[j0]; p[j0] = p[jn]; j0 = jn; }
        }
        __syncthreads();
    }

    // col_for_row: pred index per gold row
    if (p[j] != 0) colforrow[p[j] - 1] = j - 1;
    __syncthreads();

    // argsort (24 distinct values) by rank counting; write as float
    if (t < G_) {
        const int val = colforrow[t];
        int rank = 0;
        #pragma unroll
        for (int q = 0; q < G_; ++q) rank += (colforrow[q] < val) ? 1 : 0;
        row_out[b * G_ + rank] = (float)val;  // sorted pred_for_gold
        col_out[b * G_ + rank] = (float)t;    // argsort order
    }
}

// ---------------------------------------------------------------------------
extern "C" void kernel_launch(void* const* d_in, const int* in_sizes, int n_in,
                              void* d_out, int out_size, void* d_ws, size_t ws_size,
                              hipStream_t stream) {
    const float* rel = (const float*)d_in[0];
    const float* h0  = (const float*)d_in[1];
    const float* e0  = (const float*)d_in[2];
    const float* h1  = (const float*)d_in[3];
    const float* e1  = (const float*)d_in[4];
    const int* gold_rel = (const int*)d_in[5];
    const int* gold_ent = (const int*)d_in[6];

    float* out  = (float*)d_out;
    float* smax = (float*)d_ws;              // 5 * 8192
    float* ssum = smax + 5 * NROWS;          // 5 * 8192  (total 320 KB scratch)

    // 1) per-row softmax stats (read each tensor exactly once; WMMA row-sums)
    row_stats_kernel<<<dim3(NROWS / 16, 5), 32, 0, stream>>>(
        rel, h0, e0, h1, e1, smax, ssum);

    // 2) cost matrix
    cost_kernel<<<(COST_ELEMS + 255) / 256, 256, 0, stream>>>(
        rel, h0, e0, h1, e1, gold_rel, gold_ent, smax, ssum, out);

    // 3) per-batch assignment + argsort
    lsa_kernel<<<B_, N_, 0, stream>>>(
        out, out + COST_ELEMS, out + COST_ELEMS + B_ * G_);
}
